// RNNAd_6399501271154
// MI455X (gfx1250) — matrix operand order repaired and checked by reading, not compiled, not run on hardware
//
#include <hip/hip_runtime.h>
#include <hip/hip_bf16.h>
#include <math.h>

// ---------------- CDNA5 WMMA types ----------------
typedef __attribute__((ext_vector_type(16))) __bf16 v16bf;
typedef __attribute__((ext_vector_type(8)))  float  v8f;

union Frag16 {            // 16 bf16 per lane = 8 VGPRs = one WMMA A or B fragment
    v16bf v;
    uint4 q[2];
};

#define RNN_B   32
#define RNN_T   4096
#define RNN_IN  128
#define RNN_H   256
#define RNN_P   64

// ---------------- fast activations (v_tanh_f32 / v_exp_f32 / v_rcp_f32) ----------------
__device__ __forceinline__ float fast_tanh(float x) {
#if __has_builtin(__builtin_amdgcn_tanhf)
    return __builtin_amdgcn_tanhf(x);           // gfx1250 hardware v_tanh_f32
#else
    // branch-free: tanh(x) = (1 - e^(-2x)) / (1 + e^(-2x))
    float e = __builtin_amdgcn_exp2f(x * -2.885390082f);   // exp(-2x)
    return (1.f - e) * __builtin_amdgcn_rcpf(1.f + e);
#endif
}

__device__ __forceinline__ float fast_sigmoid(float x) {
    float e = __builtin_amdgcn_exp2f(x * -1.442695041f);   // exp(-x)
    return __builtin_amdgcn_rcpf(1.f + e);
}

// ---------------- f32 -> bf16 convert ----------------
__global__ __launch_bounds__(256) void cvt_bf16_kernel(const float* __restrict__ src,
                                                       __bf16* __restrict__ dst, int n) {
    int i = blockIdx.x * 256 + threadIdx.x;
    if (i < n) dst[i] = (__bf16)src[i];
}

// ---------------- generic WMMA GEMM:  C[m][n] = A[m][:] . W[n][:] + bias[n] ----------------
// Row m decomposes as (b = m&31, t = m>>5); A row ptr = A + b*sLo + t*sHi.
// mode 0: out[m*N + n] = v                      (pre-activation buffer, f32)
// mode 1: out[(b*T + t)*64 + n] = sigmoid(v)    (final FC output relayout)
template <int NTILES, int KCHUNKS>
__global__ __launch_bounds__(256) void wmma_gemm_kernel(
    const __bf16* __restrict__ A, const __bf16* __restrict__ W,
    const float* __restrict__ bias, float* __restrict__ out,
    long sLo, long sHi, int K, int mode)
{
    const int lane  = threadIdx.x & 31;
    const int wave  = threadIdx.x >> 5;              // 8 waves / block
    const int mtile = blockIdx.x * 8 + wave;         // 16 rows per wave
    const int r     = lane & 15;
    const int half  = lane >> 4;
    const int N     = NTILES * 16;

    const int m_lane = mtile * 16 + r;
    const __bf16* arow = A + (long)(m_lane & 31) * sLo + (long)(m_lane >> 5) * sHi;

    v8f acc[NTILES];
#pragma unroll
    for (int i = 0; i < NTILES; ++i) acc[i] = (v8f){0.f,0.f,0.f,0.f,0.f,0.f,0.f,0.f};

    for (int kc = 0; kc < KCHUNKS; ++kc) {
        const int k0 = kc * 32;
        // A fragment (16x32 bf16): lanes 0-15 row r hold K = k0+0..7, k0+16..23;
        // lanes 16-31 hold K = k0+8..15, k0+24..31 (ISA 16-bit A layout)
        Frag16 a;
        a.q[0] = *(const uint4*)(arow + k0 + half * 8);
        a.q[1] = *(const uint4*)(arow + k0 + 16 + half * 8);
#pragma unroll
        for (int nt = 0; nt < NTILES; ++nt) {
            // B fragment (32x16): lane (half,n) holds W[n][k0 + half*8 .. +7], W[n][k0 + 16 + half*8 .. +7]
            Frag16 b;
            const __bf16* wrow = W + (long)(nt * 16 + r) * K + k0 + half * 8;
            b.q[0] = *(const uint4*)(wrow);
            b.q[1] = *(const uint4*)(wrow + 16);
            acc[nt] = __builtin_amdgcn_wmma_f32_16x16x32_bf16(
                false, a.v, false, b.v, (short)0, acc[nt], false, false);
        }
    }

    // C layout: VGPR j -> lanes 0-15: M=j, N=lane ; lanes 16-31: M=8+j, N=lane-16
#pragma unroll
    for (int nt = 0; nt < NTILES; ++nt) {
        const int n  = nt * 16 + r;
        const float bn = bias[n];
#pragma unroll
        for (int j = 0; j < 8; ++j) {
            const int m = mtile * 16 + j + half * 8;
            const float v = acc[nt][j] + bn;
            if (mode == 0) {
                out[(long)m * N + n] = v;
            } else {
                const int t = m >> 5, bb = m & 31;
                out[(long)(bb * RNN_T + t) * RNN_P + n] = fast_sigmoid(v);
            }
        }
    }
}

// ---------------- LDS XOR swizzle: row m is 512B; swap 16B chunks by chunk^(m&15) -------
// A-fragment reads hit 16 distinct bank quads across lanes (conflict-free ds_load_b128).
__device__ __forceinline__ int hswz(int m, int byteoff) {
    return m * 512 + ((((byteoff >> 4) ^ (m & 15)) << 4) | (byteoff & 15));
}

__device__ __forceinline__ void load_a_frags(Frag16* a, const uint4* buf,
                                             int r, int half, int kc) {
    const char* hb = (const char*)buf;
#pragma unroll
    for (int mt = 0; mt < 2; ++mt) {
        const int m = mt * 16 + r;
        a[mt].q[0] = *(const uint4*)(hb + hswz(m, kc * 64 + half * 16));
        a[mt].q[1] = *(const uint4*)(hb + hswz(m, kc * 64 + half * 16 + 32));
    }
}

// ---------------- persistent RNN scan (one workgroup, W_hh fully register-resident) -------
// h_t = tanh(pre_t + h_{t-1} @ W_hh^T + b_hh), 4096 serial steps, B=32, H=256.
// 512 threads = 16 waves; wave w owns output columns [16w,16w+16): its W_hh strip lives in
// 8 B-fragments (64 VGPRs/lane). h double-buffered in LDS (bf16, XOR-swizzled, 32 KB).
__global__ __launch_bounds__(512) void rnn_scan_kernel(
    const float*  __restrict__ pre,   // [T*32][256] f32 (includes b_ih)
    const __bf16* __restrict__ Whh,   // [256][256] bf16 row-major
    const float*  __restrict__ bhh,   // [256]
    __bf16*       __restrict__ hout)  // [T*32][256] bf16
{
    __shared__ uint4 hbuf[2][RNN_B * RNN_H * 2 / 16];   // 2 x 16 KB, swizzled bf16

    const int lane = threadIdx.x & 31;
    const int wave = threadIdx.x >> 5;   // 0..15 -> N tile
    const int r    = lane & 15;
    const int half = lane >> 4;
    const int n    = wave * 16 + r;      // this lane's output column

    // Preload this wave's W_hh strip into registers (8 K-chunks x 8 VGPRs)
    Frag16 bfrag[8];
#pragma unroll
    for (int kc = 0; kc < 8; ++kc) {
        const __bf16* wrow = Whh + (long)n * RNN_H + kc * 32 + half * 8;
        bfrag[kc].q[0] = *(const uint4*)(wrow);
        bfrag[kc].q[1] = *(const uint4*)(wrow + 16);
    }
    const float bn = bhh[n];

    // h0 = 0 (swizzle-invariant)
    const uint4 z4 = {0u, 0u, 0u, 0u};
    for (int i = threadIdx.x; i < RNN_B * RNN_H * 2 / 16; i += 512) hbuf[0][i] = z4;
    __syncthreads();

    for (int t = 0; t < RNN_T; ++t) {
        const int cur = t & 1, nxt = cur ^ 1;
        const float* pre_t = pre + (long)t * RNN_B * RNN_H;

        // Issue all 16 pre loads up front; they complete under the WMMA chain.
        float pv[2][8];
#pragma unroll
        for (int mt = 0; mt < 2; ++mt)
#pragma unroll
            for (int j = 0; j < 8; ++j)
                pv[mt][j] = pre_t[(mt * 16 + j + half * 8) * RNN_H + n];

        // Two independent accumulator chains (both M-tiles) + pipelined A loads.
        v8f acc0 = (v8f){0.f,0.f,0.f,0.f,0.f,0.f,0.f,0.f};
        v8f acc1 = (v8f){0.f,0.f,0.f,0.f,0.f,0.f,0.f,0.f};
        Frag16 aCur[2], aNxt[2];
        load_a_frags(aCur, hbuf[cur], r, half, 0);
#pragma unroll
        for (int kc = 0; kc < 8; ++kc) {
            if (kc < 7) load_a_frags(aNxt, hbuf[cur], r, half, kc + 1);
            acc0 = __builtin_amdgcn_wmma_f32_16x16x32_bf16(
                false, aCur[0].v, false, bfrag[kc].v, (short)0, acc0, false, false);
            acc1 = __builtin_amdgcn_wmma_f32_16x16x32_bf16(
                false, aCur[1].v, false, bfrag[kc].v, (short)0, acc1, false, false);
            aCur[0] = aNxt[0];
            aCur[1] = aNxt[1];
        }

        // fused epilogue: + pre + b_hh, hardware tanh, bf16, write LDS(next) + global
        char* hn = (char*)hbuf[nxt];
#pragma unroll
        for (int mt = 0; mt < 2; ++mt) {
            const v8f& acc = mt ? acc1 : acc0;
#pragma unroll
            for (int j = 0; j < 8; ++j) {
                const int m = mt * 16 + j + half * 8;
                const float v = acc[j] + pv[mt][j] + bn;
                const __bf16 hb16 = (__bf16)fast_tanh(v);
                *(__bf16*)(hn + hswz(m, n * 2)) = hb16;
                hout[(long)t * RNN_B * RNN_H + m * RNN_H + n] = hb16;
            }
        }
        if (t + 1 < RNN_T) {
            __builtin_prefetch(pre + (long)(t + 1) * RNN_B * RNN_H + threadIdx.x * 16, 0, 3);
        }
        __syncthreads();   // all step-t reads done before step t+1 overwrites hbuf[cur]
    }
}

// ---------------- host-side orchestration ----------------
extern "C" void kernel_launch(void* const* d_in, const int* in_sizes, int n_in,
                              void* d_out, int out_size, void* d_ws, size_t ws_size,
                              hipStream_t stream) {
    const float* input = (const float*)d_in[0];   // [B,T,IN]
    const float* W_ih0 = (const float*)d_in[1];   // [H,IN]
    const float* W_hh0 = (const float*)d_in[2];   // [H,H]
    const float* b_ih0 = (const float*)d_in[3];
    const float* b_hh0 = (const float*)d_in[4];
    const float* W_ih1 = (const float*)d_in[5];   // [H,H]
    const float* W_hh1 = (const float*)d_in[6];   // [H,H]
    const float* b_ih1 = (const float*)d_in[7];
    const float* b_hh1 = (const float*)d_in[8];
    const float* fc_W  = (const float*)d_in[9];   // [P,H]
    const float* fc_b  = (const float*)d_in[10];
    float* out = (float*)d_out;                   // [B*T, P]

    // workspace carve-up
    char* ws = (char*)d_ws;
    size_t off = 0;
    auto carve = [&](size_t bytes) -> void* {
        void* p = ws + off;
        off += (bytes + 255) & ~(size_t)255;
        return p;
    };
    __bf16* xbf   = (__bf16*)carve((size_t)RNN_B * RNN_T * RNN_IN * 2);   // 33.5 MB
    __bf16* wih0b = (__bf16*)carve((size_t)RNN_H * RNN_IN * 2);
    __bf16* whh0b = (__bf16*)carve((size_t)RNN_H * RNN_H * 2);
    __bf16* wih1b = (__bf16*)carve((size_t)RNN_H * RNN_H * 2);
    __bf16* whh1b = (__bf16*)carve((size_t)RNN_H * RNN_H * 2);
    __bf16* fcwb  = (__bf16*)carve((size_t)RNN_P * RNN_H * 2);
    float*  pre   = (float*)carve((size_t)RNN_T * RNN_B * RNN_H * 4);     // 134 MB, reused
    __bf16* hbf   = (__bf16*)carve((size_t)RNN_T * RNN_B * RNN_H * 2);    // 67 MB, reused

    // 1. precision conversion (bf16 data path for WMMA)
    {
        int n = RNN_B * RNN_T * RNN_IN;
        cvt_bf16_kernel<<<(n + 255) / 256, 256, 0, stream>>>(input, xbf, n);
        n = RNN_H * RNN_IN;
        cvt_bf16_kernel<<<(n + 255) / 256, 256, 0, stream>>>(W_ih0, wih0b, n);
        n = RNN_H * RNN_H;
        cvt_bf16_kernel<<<(n + 255) / 256, 256, 0, stream>>>(W_hh0, whh0b, n);
        cvt_bf16_kernel<<<(n + 255) / 256, 256, 0, stream>>>(W_ih1, wih1b, n);
        cvt_bf16_kernel<<<(n + 255) / 256, 256, 0, stream>>>(W_hh1, whh1b, n);
        n = RNN_P * RNN_H;
        cvt_bf16_kernel<<<(n + 255) / 256, 256, 0, stream>>>(fc_W, fcwb, n);
    }

    const int MTILES = (RNN_B * RNN_T) / 16;     // 8192
    const int GEMM_BLOCKS = MTILES / 8;          // 1024 (8 waves/block)

    // 2. pre0 = input @ W_ih0^T + b_ih0   (A rows: b*T*IN + t*IN)
    wmma_gemm_kernel<16, 4><<<GEMM_BLOCKS, 256, 0, stream>>>(
        xbf, wih0b, b_ih0, pre,
        (long)RNN_T * RNN_IN, (long)RNN_IN, RNN_IN, 0);

    // 3. layer-0 recurrence (persistent single workgroup)
    rnn_scan_kernel<<<1, 512, 0, stream>>>(pre, whh0b, b_hh0, hbf);

    // 4. pre1 = h1 @ W_ih1^T + b_ih1     (A rows: m*H)
    wmma_gemm_kernel<16, 8><<<GEMM_BLOCKS, 256, 0, stream>>>(
        hbf, wih1b, b_ih1, pre,
        (long)RNN_H, (long)RNN_B * RNN_H, RNN_H, 0);

    // 5. layer-1 recurrence (overwrites hbf; h1 no longer needed)
    rnn_scan_kernel<<<1, 512, 0, stream>>>(pre, whh1b, b_hh1, hbf);

    // 6. out = sigmoid(h2 @ fc_W^T + fc_b), relayout [t,b] -> row b*T + t
    wmma_gemm_kernel<4, 8><<<GEMM_BLOCKS, 256, 0, stream>>>(
        hbf, fcwb, fc_b, out,
        (long)RNN_H, (long)RNN_B * RNN_H, RNN_H, 1);
}
